// SelfAttention_63247688401189
// MI455X (gfx1250) — compile-verified, matrix operand order checked by reading, not run on hardware
//
#include <hip/hip_runtime.h>
#include <hip/hip_bf16.h>

typedef __attribute__((ext_vector_type(16))) _Float16 v16h;
typedef __attribute__((ext_vector_type(8)))  _Float16 v8h;
typedef __attribute__((ext_vector_type(4)))  _Float16 v4h;
typedef __attribute__((ext_vector_type(8)))  float    v8f;
typedef __attribute__((ext_vector_type(4)))  float    v4f;

#define HID   1024
#define SEQ   2048
#define NHEAD 16
#define HDIM  64

__device__ __forceinline__ v8f wmma_f16(v16h a, v16h b, v8f c) {
    // D = A(16x32 f16) x B(32x16 f16) + C(16x16 f32)
    return __builtin_amdgcn_wmma_f32_16x16x32_f16(
        /*neg_a=*/false, a, /*neg_b=*/false, b,
        /*c_mod=*/(short)0, c, /*reuse_a=*/false, /*reuse_b=*/false);
}

__device__ __forceinline__ v8h ld8h(const _Float16* p) {  // one b128 load
    return *(const v8h*)p;
}

__device__ __forceinline__ v16h cat16(v8h a, v8h b) {
    return __builtin_shufflevector(a, b, 0,1,2,3,4,5,6,7,8,9,10,11,12,13,14,15);
}

// A/B fragment: elems 0..7 at p, elems 8..15 at p+16 (two b128 loads)
__device__ __forceinline__ v16h ldfrag(const _Float16* p) {
    return cat16(ld8h(p), ld8h(p + 16));
}

// ---------------------------------------------------------------------------
// Kernel 0: fp32 -> f16 pre-conversion (one pass; GEMM then re-reads f16).
// ---------------------------------------------------------------------------
__global__ __launch_bounds__(256)
void cvt_f16_kernel(const float* __restrict__ src, _Float16* __restrict__ dst,
                    int n4 /* element count / 4 */) {
    const int i = blockIdx.x * 256 + threadIdx.x;
    if (i < n4) {
        const v4f v = ((const v4f*)src)[i];
        v4h r;
        r[0] = (_Float16)v[0]; r[1] = (_Float16)v[1];
        r[2] = (_Float16)v[2]; r[3] = (_Float16)v[3];
        ((v4h*)dst)[i] = r;
    }
}

// ---------------------------------------------------------------------------
// Kernel 1: QKV projection.  out[n,o] = sum_k X[n,k] * W[o,k] + bias[o]
//   blockIdx.x : 128-row tile (N = B*S = 4096)
//   blockIdx.y : 128-col tile (O = 1024)
//   blockIdx.z : 0 -> Q, 1 -> K, 2 -> V (stored transposed per head)
// 8 waves (2 row x 4 col), each wave owns a 64x32 tile = 4x2 WMMA accums.
// Inner loop: 12 b128 loads + 8 WMMAs, zero conversion VALU.
// ---------------------------------------------------------------------------
__global__ __launch_bounds__(256)
void qkv_proj_kernel(const _Float16* __restrict__ Xh,
                     const _Float16* __restrict__ Wqh,
                     const _Float16* __restrict__ Wkh,
                     const _Float16* __restrict__ Wvh,
                     const float* __restrict__ bq,
                     const float* __restrict__ bk,
                     const float* __restrict__ bv,
                     _Float16* __restrict__ Qh, _Float16* __restrict__ Kh,
                     _Float16* __restrict__ Vt) {
    const int tid   = threadIdx.x;
    const int lane  = tid & 31;
    const int w     = tid >> 5;
    const int qlane = lane & 15;
    const int hi    = lane >> 4;           // 0: K set {0..7,16..23}, 1: {8..15,24..31}
    const int rw    = w >> 2;              // 0..1
    const int cw    = w & 3;               // 0..3
    const int rbase = blockIdx.x * 128 + rw * 64;
    const int cbase = blockIdx.y * 128 + cw * 32;
    const int z     = blockIdx.z;

    const _Float16* W    = (z == 0) ? Wqh : ((z == 1) ? Wkh : Wvh);
    const float*    bias = (z == 0) ? bq  : ((z == 1) ? bk  : bv);

    v8f acc[4][2] = {};

    for (int k = 0; k < HID; k += 32) {
        const int koff = k + hi * 8;
        v16h a[4], b[2];
#pragma unroll
        for (int m = 0; m < 4; ++m)
            a[m] = ldfrag(Xh + (size_t)(rbase + m * 16 + qlane) * HID + koff);
#pragma unroll
        for (int n = 0; n < 2; ++n)
            b[n] = ldfrag(W + (size_t)(cbase + n * 16 + qlane) * HID + koff);
#pragma unroll
        for (int m = 0; m < 4; ++m)
#pragma unroll
            for (int n = 0; n < 2; ++n)
                acc[m][n] = wmma_f16(a[m], b[n], acc[m][n]);
    }

    // Epilogue: bias add, cast to f16, store. Uniform z-branch hoisted.
    if (z == 2) {
        // V transposed: per (m,n) the 8 accumulator slots are contiguous in s
        // -> pack into one b128 store.
#pragma unroll
        for (int n = 0; n < 2; ++n) {
            const int col  = cbase + n * 16 + qlane;
            const int head = col >> 6, d = col & (HDIM - 1);
            const float bv_ = bias[col];
#pragma unroll
            for (int m = 0; m < 4; ++m) {
                const int row0 = rbase + m * 16 + hi * 8;   // rows row0..row0+7
                const int bi = row0 >> 11, s0 = row0 & (SEQ - 1);
                v8h pk;
#pragma unroll
                for (int j = 0; j < 8; ++j)
                    pk[j] = (_Float16)(acc[m][n][j] + bv_);
                *(v8h*)(Vt + ((size_t)(bi * NHEAD + head) * HDIM + d) * SEQ + s0) = pk;
            }
        }
    } else {
        _Float16* __restrict__ dst = (z == 0) ? Qh : Kh;
#pragma unroll
        for (int n = 0; n < 2; ++n) {
            const int col = cbase + n * 16 + qlane;
            const float bv_ = bias[col];
#pragma unroll
            for (int m = 0; m < 4; ++m) {
#pragma unroll
                for (int j = 0; j < 8; ++j) {
                    const int row = rbase + m * 16 + j + hi * 8;
                    dst[(size_t)row * HID + col] = (_Float16)(acc[m][n][j] + bv_);
                }
            }
        }
    }
}

// ---------------------------------------------------------------------------
// Kernel 2: flash attention.  One wave = 32 queries (two 16-query tiles) so
// every K/V fragment load feeds two WMMAs.
//   scoresT = K * Q^T   (M = keys, N = queries)  -> per-lane online softmax
//   ctxT    = V^T * P^T (M = dims, N = queries)  -> P^T C-layout == B-fragment
//   blockIdx.x : 256-query tile, blockIdx.y : head, blockIdx.z : batch
// Mask is pre-baked into an additive f32 bias in LDS (0 or -3e38), so the
// score epilogue is one FMA per element.
// ---------------------------------------------------------------------------
__global__ __launch_bounds__(256)
void attention_kernel(const _Float16* __restrict__ Qh,
                      const _Float16* __restrict__ Kh,
                      const _Float16* __restrict__ Vt,
                      const int* __restrict__ mask,
                      float* __restrict__ out) {
    __shared__ __attribute__((aligned(16))) float sbias[SEQ];
    const int b    = blockIdx.z;
    const int head = blockIdx.y;
    const int tid  = threadIdx.x;

    for (int i = tid; i < SEQ; i += 256)
        sbias[i] = (mask[b * SEQ + i] == 0) ? -3.0e38f : 0.0f;
    __syncthreads();

    const int lane  = tid & 31;
    const int w     = tid >> 5;
    const int qlane = lane & 15;
    const int hi    = lane >> 4;
    const int q0    = blockIdx.x * 256 + w * 32;

    // Q fragments (B-matrix of scoresT): lane = query, K = head_dim (2 chunks)
    v16h qf[2][2];
#pragma unroll
    for (int qt = 0; qt < 2; ++qt) {
        const _Float16* qp =
            Qh + (size_t)(b * SEQ + q0 + qt * 16 + qlane) * HID + head * HDIM;
#pragma unroll
        for (int kc = 0; kc < 2; ++kc)
            qf[qt][kc] = ldfrag(qp + kc * 32 + hi * 8);
    }

    const _Float16* Kbase = Kh + (size_t)b * SEQ * HID + head * HDIM;
    const _Float16* Vbase = Vt + (size_t)(b * NHEAD + head) * HDIM * SEQ;
    const float scale = 0.125f;   // 1/sqrt(64)

    float m[2] = {-1e30f, -1e30f};
    float l[2] = {0.0f, 0.0f};
    v8f acc[2][4] = {};

    for (int kt = 0; kt < SEQ; kt += 32) {
        // Prefetch next key tile's K and V lines (global_prefetch_b8).
        {
            const int ktn = (kt + 32 < SEQ) ? (kt + 32) : kt;
            __builtin_prefetch(Kbase + (size_t)(ktn + qlane) * HID + hi * 8, 0, 1);
            __builtin_prefetch(Kbase + (size_t)(ktn + 16 + qlane) * HID + hi * 8, 0, 1);
            __builtin_prefetch(Vbase + (size_t)(lane) * SEQ + ktn, 0, 1);
            __builtin_prefetch(Vbase + (size_t)(32 + lane) * SEQ + ktn, 0, 1);
        }

        // --- K A-fragments, shared by both query tiles ---
        v16h ak[2][2];   // [key sub-tile][head_dim chunk]
#pragma unroll
        for (int sub = 0; sub < 2; ++sub) {
            const _Float16* kp =
                Kbase + (size_t)(kt + sub * 16 + qlane) * HID + hi * 8;
            ak[sub][0] = ldfrag(kp);
            ak[sub][1] = ldfrag(kp + 32);
        }

        // --- additive mask bias for this lane's 16 key slots (4x ds b128) ---
        float mb0[8], mb1[8];
        {
            const v4f* bp0 = (const v4f*)&sbias[kt + hi * 8];
            const v4f* bp1 = (const v4f*)&sbias[kt + 16 + hi * 8];
            v4f a0 = bp0[0], a1 = bp0[1], c0 = bp1[0], c1 = bp1[1];
#pragma unroll
            for (int j = 0; j < 4; ++j) {
                mb0[j] = a0[j]; mb0[4 + j] = a1[j];
                mb1[j] = c0[j]; mb1[4 + j] = c1[j];
            }
        }

        v16h pb[2];   // P^T B-fragments, one per query tile
#pragma unroll
        for (int qt = 0; qt < 2; ++qt) {
            // scoresT: two 16(key) x 16(query) tiles
            v8f st[2];
#pragma unroll
            for (int sub = 0; sub < 2; ++sub) {
                v8f c = {};
                c = wmma_f16(ak[sub][0], qf[qt][0], c);
                c = wmma_f16(ak[sub][1], qf[qt][1], c);
                st[sub] = c;
            }

            // scale+mask via single FMA, then per-query online softmax
            float p0[8], p1[8];
            float mx = -1e30f;
#pragma unroll
            for (int j = 0; j < 8; ++j) {
                const float s0 = fmaf(st[0][j], scale, mb0[j]);
                const float s1 = fmaf(st[1][j], scale, mb1[j]);
                p0[j] = s0; p1[j] = s1;
                mx = fmaxf(mx, fmaxf(s0, s1));
            }
            mx = fmaxf(mx, __shfl_xor(mx, 16, 32));  // combine half-rows
            const float mnew   = fmaxf(m[qt], mx);
            const float escale = __expf(m[qt] - mnew);

            float rs = 0.0f;
            v16h p;
#pragma unroll
            for (int j = 0; j < 8; ++j) {
                const float e0 = __expf(p0[j] - mnew);
                const float e1 = __expf(p1[j] - mnew);
                rs += e0 + e1;
                p[j]     = (_Float16)e0;
                p[8 + j] = (_Float16)e1;
            }
            rs += __shfl_xor(rs, 16, 32);
            l[qt] = l[qt] * escale + rs;
            m[qt] = mnew;
            pb[qt] = p;

            // rescale accumulators for this query tile
#pragma unroll
            for (int dt = 0; dt < 4; ++dt)
#pragma unroll
                for (int j = 0; j < 8; ++j) acc[qt][dt][j] *= escale;
        }

        // --- ctxT += V^T * P^T : each V fragment feeds both query tiles ---
#pragma unroll
        for (int dt = 0; dt < 4; ++dt) {
            const _Float16* vp =
                Vbase + (size_t)(dt * 16 + qlane) * SEQ + kt + hi * 8;
            v16h av = ldfrag(vp);
            acc[0][dt] = wmma_f16(av, pb[0], acc[0][dt]);
            acc[1][dt] = wmma_f16(av, pb[1], acc[1][dt]);
        }
    }

    // --- write out: out[b, q, head*64 + d]; 8 floats per (qt,dt) are
    //     contiguous in d -> two b128 stores each ---
#pragma unroll
    for (int qt = 0; qt < 2; ++qt) {
        const float linv = 1.0f / l[qt];
        float* op = out + (size_t)(b * SEQ + q0 + qt * 16 + qlane) * HID
                        + head * HDIM + hi * 8;
#pragma unroll
        for (int dt = 0; dt < 4; ++dt) {
            v4f lo, hic;
#pragma unroll
            for (int j = 0; j < 4; ++j) {
                lo[j]  = acc[qt][dt][j]     * linv;
                hic[j] = acc[qt][dt][4 + j] * linv;
            }
            *(v4f*)(op + dt * 16)     = lo;
            *(v4f*)(op + dt * 16 + 4) = hic;
        }
    }
}

// ---------------------------------------------------------------------------

extern "C" void kernel_launch(void* const* d_in, const int* in_sizes, int n_in,
                              void* d_out, int out_size, void* d_ws, size_t ws_size,
                              hipStream_t stream) {
    (void)in_sizes; (void)n_in; (void)out_size; (void)ws_size;
    const float* X    = (const float*)d_in[0];
    const int*   mask = (const int*)d_in[1];
    const float* Wq   = (const float*)d_in[2];
    const float* bq   = (const float*)d_in[3];
    const float* Wk   = (const float*)d_in[4];
    const float* bk   = (const float*)d_in[5];
    const float* Wv   = (const float*)d_in[6];
    const float* bv   = (const float*)d_in[7];
    float* out = (float*)d_out;

    const size_t nX = (size_t)2 * SEQ * HID;      // 4096*1024
    const size_t nW = (size_t)HID * HID;          // 1024*1024
    _Float16* Qh  = (_Float16*)d_ws;
    _Float16* Kh  = Qh  + nX;
    _Float16* Vt  = Kh  + nX;
    _Float16* Xh  = Vt  + nX;
    _Float16* Wqh = Xh  + nX;
    _Float16* Wkh = Wqh + nW;
    _Float16* Wvh = Wkh + nW;                     // total 38 MiB workspace

    // Pass 0: one-shot fp32 -> f16 conversion of X and weights.
    cvt_f16_kernel<<<(int)(nX / 4 / 256), 256, 0, stream>>>(X,  Xh,  (int)(nX / 4));
    cvt_f16_kernel<<<(int)(nW / 4 / 256), 256, 0, stream>>>(Wq, Wqh, (int)(nW / 4));
    cvt_f16_kernel<<<(int)(nW / 4 / 256), 256, 0, stream>>>(Wk, Wkh, (int)(nW / 4));
    cvt_f16_kernel<<<(int)(nW / 4 / 256), 256, 0, stream>>>(Wv, Wvh, (int)(nW / 4));

    dim3 g1(4096 / 128, HID / 128, 3);            // 32 x 8 x 3
    qkv_proj_kernel<<<g1, 256, 0, stream>>>(Xh, Wqh, Wkh, Wvh, bq, bk, bv,
                                            Qh, Kh, Vt);

    dim3 g2(SEQ / 256, NHEAD, 2);                 // 8 x 16 x 2
    attention_kernel<<<g2, 256, 0, stream>>>(Qh, Kh, Vt, mask, out);
}
